// Multihead_Self_Attention_63780264346139
// MI455X (gfx1250) — matrix operand order repaired and checked by reading, not compiled, not run on hardware
//
#include <hip/hip_runtime.h>
#include <hip/hip_bf16.h>

// ---------------------------------------------------------------------------
// Multi-head self-attention with RoPE, bf16 WMMA pipeline for gfx1250 (MI455X)
//   B=2, S=2048, D_MODEL=1024, H=16, D_HEAD=64.  wave32, V_WMMA_F32_16X16X32_BF16.
//   GEMM loops use explicit ping-pong operand buffers (no register rotation)
//   so each WMMA consumes tiles loaded a half-iteration earlier while the
//   other buffer's load clause is in flight.
// ---------------------------------------------------------------------------

#define BATCH   2
#define SEQ     2048
#define DMODEL  1024
#define NHEADS  16
#define DHEAD   64
#define ROWS    (BATCH * SEQ)        // 4096

typedef __bf16 v16bf __attribute__((ext_vector_type(16)));
typedef __bf16 v8bf  __attribute__((ext_vector_type(8)));
typedef float  v8f   __attribute__((ext_vector_type(8)));

// D = A(16x32) * B(32x16) + C, bf16 in / f32 out
static __device__ __forceinline__ v8f wmma_bf16(v16bf a, v16bf b, v8f c) {
    return __builtin_amdgcn_wmma_f32_16x16x32_bf16(
        /*neg_a=*/false, a, /*neg_b=*/false, b,
        /*c_mod=*/(short)0, c, /*reuse_a=*/false, /*reuse_b=*/false);
}

// A-operand (16x32 bf16) from a row-major row pointer. Contiguous k order:
// lane half h: element t -> k = 16h+t.
static __device__ __forceinline__ v16bf load_A_row(const __bf16* rowbase, int kb, int h) {
    const v8bf lo = *(const v8bf*)(rowbase + kb * 32 + 16 * h);
    const v8bf hi = *(const v8bf*)(rowbase + kb * 32 + 16 * h + 8);
    return __builtin_shufflevector(lo, hi, 0,1,2,3,4,5,6,7,8,9,10,11,12,13,14,15);
}

// ------------------------------- convert -----------------------------------
// 8 f32 -> 8 bf16 per thread: 2x global_load_b128 + 1x global_store_b128.
__global__ __launch_bounds__(256) void k_f32_to_bf16v(const float4* __restrict__ in,
                                                      v8bf* __restrict__ out, int n8) {
    int i = blockIdx.x * blockDim.x + threadIdx.x;
    if (i < n8) {
        float4 x0 = in[2 * i];
        float4 x1 = in[2 * i + 1];
        v8bf o;
        o[0] = (__bf16)x0.x; o[1] = (__bf16)x0.y; o[2] = (__bf16)x0.z; o[3] = (__bf16)x0.w;
        o[4] = (__bf16)x1.x; o[5] = (__bf16)x1.y; o[6] = (__bf16)x1.z; o[7] = (__bf16)x1.w;
        out[i] = o;
    }
}

// ----------------------- fused QKV projection + RoPE -----------------------
// C[m,e] = sum_d X[m,d] * W[e,d]  (einsum 'bsd,ed->bse').
// A from row-major Xbf rows; B lane n=e reads W row e (contiguous k=d).
__global__ __launch_bounds__(256) void k_proj_qkv(
        const __bf16* __restrict__ Xbf,
        const __bf16* __restrict__ Wq, const __bf16* __restrict__ Wk,
        const __bf16* __restrict__ Wv, const int* __restrict__ tp,
        __bf16* __restrict__ Qo,  // [b,h,s,dh] row-major, pre-scaled by 1/8
        __bf16* __restrict__ Ko,  // [b,h,s,dh] row-major
        __bf16* __restrict__ Vt)  // [b,h,dh,s] (transposed)
{
    const int tid  = threadIdx.x;
    const int widx = tid >> 5;
    const int lane = tid & 31;
    const int ln   = lane & 15;
    const int h    = lane >> 4;
    const int w    = blockIdx.x * 8 + widx;   // 16384 waves
    const int mt   = w >> 6;                  // 0..255 row tiles
    const int nt   = w & 63;                  // 0..63  e tiles

    const __bf16* arow = Xbf + (size_t)(mt * 16 + ln) * DMODEL;
    const __bf16* bq = Wq + (size_t)(nt * 16 + ln) * DMODEL + 16 * h;
    const __bf16* bk = Wk + (size_t)(nt * 16 + ln) * DMODEL + 16 * h;
    const __bf16* bv = Wv + (size_t)(nt * 16 + ln) * DMODEL + 16 * h;

    v8f cq = {}, ck = {}, cv = {};

    // ping-pong operand buffers (no rotation copies)
    v16bf a0  = load_A_row(arow, 0, h);
    v16bf Bq0 = *(const v16bf*)(bq);
    v16bf Bk0 = *(const v16bf*)(bk);
    v16bf Bv0 = *(const v16bf*)(bv);
    v16bf a1, Bq1, Bk1, Bv1;

#pragma unroll 2
    for (int i = 0; i < DMODEL / 64; ++i) {       // 16 iters, 2 k-blocks each
        const int kb = 2 * i;
        a1  = load_A_row(arow, kb + 1, h);
        Bq1 = *(const v16bf*)(bq + (kb + 1) * 32);
        Bk1 = *(const v16bf*)(bk + (kb + 1) * 32);
        Bv1 = *(const v16bf*)(bv + (kb + 1) * 32);
        cq = wmma_bf16(a0, Bq0, cq);
        ck = wmma_bf16(a0, Bk0, ck);
        cv = wmma_bf16(a0, Bv0, cv);
        if (i + 1 < DMODEL / 64) {
            a0  = load_A_row(arow, kb + 2, h);
            Bq0 = *(const v16bf*)(bq + (kb + 2) * 32);
            Bk0 = *(const v16bf*)(bk + (kb + 2) * 32);
            Bv0 = *(const v16bf*)(bv + (kb + 2) * 32);
        }
        cq = wmma_bf16(a1, Bq1, cq);
        ck = wmma_bf16(a1, Bk1, ck);
        cv = wmma_bf16(a1, Bv1, cv);
    }

    // RoPE on C-layout: lane holds column e = nt*16+ln, rows r+8h in VGPR r.
    const int e  = nt * 16 + ln;
    const int dh = e & (DHEAD - 1);
    const int hd = e >> 6;                       // head index
    const int i2 = dh >> 1;
    // inv_freq = 10000^(-(2i)/64) = exp2(-(2i/64)*log2(10000))
    const float invf = exp2f(-(float)(2 * i2) * (13.287712379549449f / 64.0f));
    const float sgn  = (dh & 1) ? 1.0f : -1.0f; // even: -odd*sin, odd: +even*sin
    const int rowbase = mt * 16 + 8 * h;

#pragma unroll
    for (int r = 0; r < 8; ++r) {
        const int row = rowbase + r;
        const int b   = row >> 11;               // /SEQ
        const int s   = row & (SEQ - 1);
        const float pos = (float)tp[s];
        const float ang = pos * invf;
        const float cc = __cosf(ang);
        const float ss = __sinf(ang);
        const float pq = __shfl_xor(cq[r], 1, 32);   // partner e^1 value
        const float pk = __shfl_xor(ck[r], 1, 32);
        const float qv = (cq[r] * cc + sgn * pq * ss) * 0.125f; // fold 1/sqrt(64)
        const float kv = ck[r] * cc + sgn * pk * ss;
        const size_t idx = ((size_t)(b * NHEADS + hd) * SEQ + s) * DHEAD + dh;
        Qo[idx] = (__bf16)qv;
        Ko[idx] = (__bf16)kv;
        Vt[((size_t)(b * NHEADS + hd) * DHEAD + dh) * SEQ + s] = (__bf16)cv[r];
    }
}

// --------------------------- flash attention -------------------------------
// One wave = one (b,h,16-row q tile). Causal kv chunks of 32.
// Pipelined: V tiles + next-chunk K tiles load during softmax VALU work.
__global__ __launch_bounds__(256) void k_attn(
        const __bf16* __restrict__ Q, const __bf16* __restrict__ K,
        const __bf16* __restrict__ Vt, __bf16* __restrict__ AO)
{
    __shared__ __align__(32) __bf16 plds[8 * 512];   // 8 waves x 16x32 bf16

    const int tid  = threadIdx.x;
    const int widx = tid >> 5;
    const int lane = tid & 31;
    const int ln   = lane & 15;
    const int h    = lane >> 4;
    const int w    = blockIdx.x * 8 + widx;   // 4096 waves
    const int bh   = w >> 7;                  // 0..31 (b*16+h)
    const int qt   = w & 127;                 // q tile within head
    const int qb   = qt * 16;

    const __bf16* qrow  = Q + ((size_t)bh * SEQ + qb + ln) * DHEAD;
    const v16bf a0 = load_A_row(qrow, 0, h);  // d 0..31
    const v16bf a1 = load_A_row(qrow, 1, h);  // d 32..63

    v8f o0 = {}, o1 = {}, o2 = {}, o3 = {};
    float mr[8], lr[8];
#pragma unroll
    for (int r = 0; r < 8; ++r) { mr[r] = -3.0e38f; lr[r] = 0.0f; }

    const __bf16* kbase = K  + (size_t)bh * SEQ * DHEAD;
    const __bf16* vbase = Vt + (size_t)bh * DHEAD * SEQ;
    __bf16* pw = plds + widx * 512;
    const int nch = (qt >> 1) + 1;

    // prologue: K tiles for chunk 0
    const __bf16* kp0 = kbase + (size_t)ln * DHEAD + 16 * h;
    const __bf16* kp1 = kbase + (size_t)(16 + ln) * DHEAD + 16 * h;
    v16bf B00 = *(const v16bf*)(kp0);
    v16bf B01 = *(const v16bf*)(kp0 + 32);
    v16bf B10 = *(const v16bf*)(kp1);
    v16bf B11 = *(const v16bf*)(kp1 + 32);

    for (int c = 0; c < nch; ++c) {
        const int kvb = c * 32;
        // ---- scores: S[16q x 32kv], two 16-col WMMA tiles, K-dim = dhead ----
        v8f s0 = {}, s1 = {};
        s0 = wmma_bf16(a0, B00, s0); s0 = wmma_bf16(a1, B01, s0);
        s1 = wmma_bf16(a0, B10, s1); s1 = wmma_bf16(a1, B11, s1);

        // ---- V tiles for this chunk: independent of softmax, load now ----
        const __bf16* vp = vbase + (size_t)ln * SEQ + kvb + 16 * h;
        v16bf Bv0 = *(const v16bf*)(vp);
        v16bf Bv1 = *(const v16bf*)(vp + 16 * SEQ);
        v16bf Bv2 = *(const v16bf*)(vp + 32 * SEQ);
        v16bf Bv3 = *(const v16bf*)(vp + 48 * SEQ);

        // ---- next-chunk K tiles: load during softmax ----
        if (c + 1 < nch) {
            const __bf16* nk0 = kbase + (size_t)(kvb + 32 + ln)      * DHEAD + 16 * h;
            const __bf16* nk1 = kbase + (size_t)(kvb + 48 + ln)      * DHEAD + 16 * h;
            B00 = *(const v16bf*)(nk0);
            B01 = *(const v16bf*)(nk0 + 32);
            B10 = *(const v16bf*)(nk1);
            B11 = *(const v16bf*)(nk1 + 32);
            if (c + 2 < nch) __builtin_prefetch(nk0 + 32 * DHEAD, 0, 0);
        }

        // ---- causal mask + online softmax (rows across 16 lanes) ----
        const int col0 = kvb + ln, col1 = kvb + 16 + ln;
#pragma unroll
        for (int r = 0; r < 8; ++r) {
            const int qi = qb + r + 8 * h;
            float v0 = (col0 <= qi) ? s0[r] : -1.0e30f;
            float v1 = (col1 <= qi) ? s1[r] : -1.0e30f;
            float mx = fmaxf(v0, v1);
            mx = fmaxf(mx, __shfl_xor(mx, 1, 32));
            mx = fmaxf(mx, __shfl_xor(mx, 2, 32));
            mx = fmaxf(mx, __shfl_xor(mx, 4, 32));
            mx = fmaxf(mx, __shfl_xor(mx, 8, 32));
            const float mnew  = fmaxf(mr[r], mx);
            const float alpha = __expf(mr[r] - mnew);
            const float p0 = __expf(v0 - mnew);
            const float p1 = __expf(v1 - mnew);
            float sm = p0 + p1;
            sm += __shfl_xor(sm, 1, 32);
            sm += __shfl_xor(sm, 2, 32);
            sm += __shfl_xor(sm, 4, 32);
            sm += __shfl_xor(sm, 8, 32);
            lr[r] = lr[r] * alpha + sm;
            mr[r] = mnew;
            o0[r] = o0[r] * alpha; o1[r] = o1[r] * alpha;
            o2[r] = o2[r] * alpha; o3[r] = o3[r] * alpha;
            pw[(r + 8 * h) * 32 + ln]      = (__bf16)p0;
            pw[(r + 8 * h) * 32 + 16 + ln] = (__bf16)p1;
        }
        asm volatile("s_wait_dscnt 0" ::: "memory");   // LDS writes visible

        // ---- read P back in A-layout (16q x 32kv) ----
        const v8bf plo = *(const v8bf*)&pw[ln * 32 + 16 * h];
        const v8bf phi = *(const v8bf*)&pw[ln * 32 + 16 * h + 8];
        const v16bf ap = __builtin_shufflevector(plo, phi,
                              0,1,2,3,4,5,6,7,8,9,10,11,12,13,14,15);

        // ---- O += P * V ----
        o0 = wmma_bf16(ap, Bv0, o0);
        o1 = wmma_bf16(ap, Bv1, o1);
        o2 = wmma_bf16(ap, Bv2, o2);
        o3 = wmma_bf16(ap, Bv3, o3);
    }

    // ---- normalize and store to AttnOut [b,s, h*64+d] bf16 ----
    const int b  = bh >> 4;
    const int hh = bh & 15;
#pragma unroll
    for (int r = 0; r < 8; ++r) {
        const float inv = 1.0f / lr[r];
        const int qi = qb + r + 8 * h;
        __bf16* dst = AO + ((size_t)(b * SEQ + qi)) * DMODEL + hh * DHEAD;
        dst[0 * 16 + ln] = (__bf16)(o0[r] * inv);
        dst[1 * 16 + ln] = (__bf16)(o1[r] * inv);
        dst[2 * 16 + ln] = (__bf16)(o2[r] * inv);
        dst[3 * 16 + ln] = (__bf16)(o3[r] * inv);
    }
}

// --------------------------- output projection -----------------------------
// out[m,e] = sum_d AO[m,d] * Wo[e,d] ; 4 N-tiles per wave, ping-pong k loop.
__global__ __launch_bounds__(256) void k_oproj(
        const __bf16* __restrict__ AObf, const __bf16* __restrict__ Wo,
        float* __restrict__ out)
{
    const int tid  = threadIdx.x;
    const int widx = tid >> 5;
    const int lane = tid & 31;
    const int ln   = lane & 15;
    const int h    = lane >> 4;
    const int w    = blockIdx.x * 8 + widx;   // 4096 waves
    const int mt   = w >> 4;                  // 0..255
    const int ng   = w & 15;                  // 0..15 (64 cols each)

    const __bf16* arow = AObf + (size_t)(mt * 16 + ln) * DMODEL;
    const __bf16* wr0 = Wo + (size_t)(ng * 64 +  0 + ln) * DMODEL + 16 * h;
    const __bf16* wr1 = Wo + (size_t)(ng * 64 + 16 + ln) * DMODEL + 16 * h;
    const __bf16* wr2 = Wo + (size_t)(ng * 64 + 32 + ln) * DMODEL + 16 * h;
    const __bf16* wr3 = Wo + (size_t)(ng * 64 + 48 + ln) * DMODEL + 16 * h;

    v8f c0 = {}, c1 = {}, c2 = {}, c3 = {};

    v16bf a0  = load_A_row(arow, 0, h);
    v16bf P0  = *(const v16bf*)(wr0);
    v16bf P1  = *(const v16bf*)(wr1);
    v16bf P2  = *(const v16bf*)(wr2);
    v16bf P3  = *(const v16bf*)(wr3);
    v16bf a1, Q0, Q1, Q2, Q3;

#pragma unroll 2
    for (int i = 0; i < DMODEL / 64; ++i) {       // 16 iters, 2 k-blocks each
        const int kb = 2 * i;
        a1 = load_A_row(arow, kb + 1, h);
        Q0 = *(const v16bf*)(wr0 + (kb + 1) * 32);
        Q1 = *(const v16bf*)(wr1 + (kb + 1) * 32);
        Q2 = *(const v16bf*)(wr2 + (kb + 1) * 32);
        Q3 = *(const v16bf*)(wr3 + (kb + 1) * 32);
        c0 = wmma_bf16(a0, P0, c0);
        c1 = wmma_bf16(a0, P1, c1);
        c2 = wmma_bf16(a0, P2, c2);
        c3 = wmma_bf16(a0, P3, c3);
        if (i + 1 < DMODEL / 64) {
            a0 = load_A_row(arow, kb + 2, h);
            P0 = *(const v16bf*)(wr0 + (kb + 2) * 32);
            P1 = *(const v16bf*)(wr1 + (kb + 2) * 32);
            P2 = *(const v16bf*)(wr2 + (kb + 2) * 32);
            P3 = *(const v16bf*)(wr3 + (kb + 2) * 32);
        }
        c0 = wmma_bf16(a1, Q0, c0);
        c1 = wmma_bf16(a1, Q1, c1);
        c2 = wmma_bf16(a1, Q2, c2);
        c3 = wmma_bf16(a1, Q3, c3);
    }

#pragma unroll
    for (int r = 0; r < 8; ++r) {
        const int row = mt * 16 + r + 8 * h;
        float* dst = out + (size_t)row * DMODEL + ng * 64 + ln;
        dst[ 0] = c0[r];
        dst[16] = c1[r];
        dst[32] = c2[r];
        dst[48] = c3[r];
    }
}

// ------------------------------- launcher ----------------------------------
extern "C" void kernel_launch(void* const* d_in, const int* in_sizes, int n_in,
                              void* d_out, int out_size, void* d_ws, size_t ws_size,
                              hipStream_t stream) {
    const float* X  = (const float*)d_in[0];
    const int*   tp = (const int*)  d_in[1];
    const float* wq = (const float*)d_in[2];
    const float* wk = (const float*)d_in[3];
    const float* wv = (const float*)d_in[4];
    const float* wo = (const float*)d_in[5];
    float* out = (float*)d_out;

    // workspace carve-up (bf16 staging, 48 MB total)
    char* ws = (char*)d_ws;
    __bf16* Xbf = (__bf16*)(ws);                                   // 8 MB
    __bf16* Wqb = (__bf16*)(ws + (8u  << 20));                     // 2 MB each
    __bf16* Wkb = (__bf16*)(ws + (10u << 20));
    __bf16* Wvb = (__bf16*)(ws + (12u << 20));
    __bf16* Wob = (__bf16*)(ws + (14u << 20));
    __bf16* Qbf = (__bf16*)(ws + (16u << 20));                     // 8 MB
    __bf16* Kbf = (__bf16*)(ws + (24u << 20));                     // 8 MB
    __bf16* Vtb = (__bf16*)(ws + (32u << 20));                     // 8 MB
    __bf16* AOb = (__bf16*)(ws + (40u << 20));                     // 8 MB

    const int NX8 = ROWS * DMODEL / 8;       // 524288
    const int NW8 = DMODEL * DMODEL / 8;     // 131072

    k_f32_to_bf16v<<<NX8 / 256, 256, 0, stream>>>((const float4*)X,  (v8bf*)Xbf, NX8);
    k_f32_to_bf16v<<<NW8 / 256, 256, 0, stream>>>((const float4*)wq, (v8bf*)Wqb, NW8);
    k_f32_to_bf16v<<<NW8 / 256, 256, 0, stream>>>((const float4*)wk, (v8bf*)Wkb, NW8);
    k_f32_to_bf16v<<<NW8 / 256, 256, 0, stream>>>((const float4*)wv, (v8bf*)Wvb, NW8);
    k_f32_to_bf16v<<<NW8 / 256, 256, 0, stream>>>((const float4*)wo, (v8bf*)Wob, NW8);

    // 256 m-tiles * 64 n-tiles = 16384 waves -> 2048 blocks of 8 waves
    k_proj_qkv<<<2048, 256, 0, stream>>>(Xbf, Wqb, Wkb, Wvb, tp, Qbf, Kbf, Vtb);

    // 32 (b,h) * 128 q-tiles = 4096 waves -> 512 blocks
    k_attn<<<512, 256, 0, stream>>>(Qbf, Kbf, Vtb, AOb);

    // 256 m-tiles * 16 n-groups = 4096 waves -> 512 blocks
    k_oproj<<<512, 256, 0, stream>>>(AOb, Wob, out);
}
